// VisionMamba_87213605913197
// MI455X (gfx1250) — matrix operand order repaired and checked by reading, not compiled, not run on hardware
//
#include <hip/hip_runtime.h>

// ---------------------------------------------------------------------------
// Vision Mamba forward on gfx1250 (MI455X, wave32, WMMA, async LDS DMA).
//
// Design:
//  * All GEMM operands pre-converted to fp16 scratch, padded to tile
//    multiples (M:128, N:64, K:32) -> zero load-side guards in the GEMM.
//  * B matrices stored n-major (transposed) so every 16-byte global chunk is
//    K-contiguous and maps to one contiguous fragment slot in LDS.
//  * LDS holds tiles in WMMA *fragment order*: per lane 16 contiguous halves
//    -> operand build is two ds_load_b128, no scalar gathers.
//  * Staging uses GLOBAL_LOAD_ASYNC_TO_LDS_B128, double-buffered: next tile's
//    DMA is issued before draining the current one with a partial
//    s_wait_asynccnt(3) (3 async ops in flight per thread, in-order
//    completion) -> DMA overlaps WMMA.
//  * Scan / norms / conv stay fp32 VALU; WMMA accumulates fp32.
// ---------------------------------------------------------------------------

constexpr int kB = 32, kC = 3, kIMG = 224, kP = 16;
constexpr int kD = 384, kDEPTH = 12, kN = 16, kK = 4;
constexpr int kDI = 2 * kD;                    // 768
constexpr int kR = (kD + 15) / 16;             // 24
constexpr int kL = (kIMG / kP) * (kIMG / kP);  // 196
constexpr int kNC = 1000;
constexpr int kTOK = kB * kL;                  // 6272 = 49 * 128
constexpr int kXZ = 2 * kDI;                   // 1536
constexpr int kDBC = kR + 2 * kN;              // 56

typedef __attribute__((ext_vector_type(16))) _Float16     v16h;
typedef __attribute__((ext_vector_type(8)))  float        v8f;
typedef __attribute__((ext_vector_type(8)))  unsigned int u8i;
typedef __attribute__((ext_vector_type(4)))  int          v4i;

#define TM 128
#define TN 64
#define TK 32

#if defined(__gfx1250__) && __has_builtin(__builtin_amdgcn_global_load_async_to_lds_b128) && __has_builtin(__builtin_amdgcn_s_wait_asynccnt)
#define USE_ASYNC_LDS 1
typedef __attribute__((address_space(1))) v4i gv4i;   // global int4
typedef __attribute__((address_space(3))) v4i lv4i;   // LDS int4
#else
#define USE_ASYNC_LDS 0
#endif

__device__ __forceinline__ void copy16B_to_lds(const _Float16* g, _Float16* l)
{
#if USE_ASYNC_LDS
    __builtin_amdgcn_global_load_async_to_lds_b128(
        (gv4i*)(void*)const_cast<_Float16*>(g), (lv4i*)(void*)l, 0, 0);
#else
    *(uint4*)l = *(const uint4*)g;
#endif
}

// ---------------------------------------------------------------------------
// GEMM: C[M,N] (+bias)(+beta*C) = A[M,K]fp16 @ B[K,N] (B stored n-major fp16).
// Operands fully padded: no load guards. Block tile 128x64, K-step 32,
// 256 threads = 8 waves; wave w -> m-tile w, all 4 n-tiles (A reuse, 4 WMMA).
// Double-buffered LDS: async DMA for tile s+1 overlaps WMMA on tile s.
// ---------------------------------------------------------------------------
__global__ __launch_bounds__(256) void gemm_f16_wmma_kernel(
    const _Float16* __restrict__ A,   // [Mpad][lda]
    const _Float16* __restrict__ Bt,  // [Npad][ldb] n-major
    float* __restrict__ C, _Float16* __restrict__ C16,
    const float* __restrict__ bias,
    int M, int N, int K, int lda, int ldb, int ldc, int ldc16, int beta)
{
    // fragment-order LDS: [buf][tile][lane][16 halves contiguous]
    __shared__ __attribute__((aligned(32))) _Float16 Afrag[2][8][32][16]; // 16 KB
    __shared__ __attribute__((aligned(32))) _Float16 Bfrag[2][4][32][16]; //  8 KB

    const int tid  = threadIdx.x;
    const int wave = tid >> 5;
    const int lane = tid & 31;
    const int m0 = blockIdx.x * TM;
    const int n0 = blockIdx.y * TN;

    // per-thread staging coordinates (3 async 16B chunks per tile)
    // A: chunks tid, tid+256 of 512; layout lane = m + 16*((k>>3)&1),
    //    i = (k&7) + 8*(k>>4)
    const int ar0 = tid >> 2,            ak0 = (tid & 3) * 8;
    const int ar1 = (tid + 256) >> 2,    ak1 = ak0;   // same kc pattern
    // B: chunk tid of 256; layout lane = n + 16*((k>>3)&1), i = (k&7) + 8*(k>>4)
    const int bnr = tid >> 2,            bkc = (tid & 3) * 8;

    auto stage = [&](int kb, int buf) {
        {
            const int mt = ar0 >> 4, m = ar0 & 15;
            const int sel = (ak0 >> 3) & 1, iblk = ak0 >> 4;
            copy16B_to_lds(A + (size_t)(m0 + ar0) * lda + kb + ak0,
                           &Afrag[buf][mt][m + 16 * sel][iblk * 8]);
        }
        {
            const int mt = ar1 >> 4, m = ar1 & 15;
            const int sel = (ak1 >> 3) & 1, iblk = ak1 >> 4;
            copy16B_to_lds(A + (size_t)(m0 + ar1) * lda + kb + ak1,
                           &Afrag[buf][mt][m + 16 * sel][iblk * 8]);
        }
        {
            const int nt = bnr >> 4, nl = bnr & 15;
            const int sel = (bkc >> 3) & 1, i0 = (bkc >> 4) * 8;
            copy16B_to_lds(Bt + (size_t)(n0 + bnr) * ldb + kb + bkc,
                           &Bfrag[buf][nt][nl + 16 * sel][i0]);
        }
    };

    v8f acc[4] = {{}, {}, {}, {}};

    const int ksteps = K / TK;           // K is always a multiple of 32
    stage(0, 0);
    for (int s = 0; s < ksteps; ++s) {
        const int buf = s & 1;
        const bool more = (s + 1) < ksteps;
        if (more) stage((s + 1) * TK, buf ^ 1);   // overlap DMA with compute
#if USE_ASYNC_LDS
        if (more) __builtin_amdgcn_s_wait_asynccnt(3);  // drain tile s only
        else      __builtin_amdgcn_s_wait_asynccnt(0);
#endif
        __syncthreads();

        // ---- operand build: 32B contiguous LDS loads per fragment ----
        const v16h a = __builtin_bit_cast(
            v16h, *(const u8i*)&Afrag[buf][wave][lane][0]);
        #pragma unroll
        for (int j = 0; j < 4; ++j) {
            const v16h b = __builtin_bit_cast(
                v16h, *(const u8i*)&Bfrag[buf][j][lane][0]);
            acc[j] = __builtin_amdgcn_wmma_f32_16x16x32_f16(
                false, a, false, b, (short)0, acc[j], false, false);
        }
        __syncthreads();   // protect buf^1 reuse by next stage()
    }

    // ---- epilogue: D layout m = r + 8*(lane>>4), n = lane&15 ----
    const int lm = lane & 15, sel = lane >> 4;
    #pragma unroll
    for (int j = 0; j < 4; ++j) {
        const int gn = n0 + j * 16 + lm;
        if (gn >= N) continue;
        const float bv = bias ? bias[gn] : 0.0f;
        #pragma unroll
        for (int r = 0; r < 8; ++r) {
            const int gm = m0 + wave * 16 + sel * 8 + r;
            if (gm >= M) continue;
            float v = acc[j][r] + bv;
            if (beta) v += C[(size_t)gm * ldc + gn];
            C[(size_t)gm * ldc + gn] = v;
            if (C16) C16[(size_t)gm * ldc16 + gn] = (_Float16)v;
        }
    }
}

// ---------------------------------------------------------------------------
// Converters (run once per call; weights amortized over 49 M-blocks)
// ---------------------------------------------------------------------------
// dst[n][k] (n-major, padded, zero-filled) from src[k][n] row-major fp32
__global__ __launch_bounds__(256) void convert_transpose_f16_kernel(
    const float* __restrict__ src, _Float16* __restrict__ dst,
    int K, int N, int Kpad, int Npad)
{
    const int idx = blockIdx.x * blockDim.x + threadIdx.x;
    if (idx >= Npad * Kpad) return;
    const int n = idx / Kpad, k = idx % Kpad;
    const float v = (n < N && k < K) ? src[(size_t)k * N + n] : 0.0f;
    dst[idx] = (_Float16)v;
}

__global__ __launch_bounds__(256) void convert_copy_f16_kernel(
    const float* __restrict__ src, _Float16* __restrict__ dst, long n)
{
    const long idx = (long)blockIdx.x * blockDim.x + threadIdx.x;
    if (idx < n) dst[idx] = (_Float16)src[idx];
}

// im2col directly to fp16 A panel: A0[b*L+l][c*256+iy*16+ix]
__global__ __launch_bounds__(256) void im2col_f16_kernel(
    const float* __restrict__ x, _Float16* __restrict__ A0)
{
    const int idx = blockIdx.x * blockDim.x + threadIdx.x;
    if (idx >= kTOK * (kC * kP * kP)) return;
    const int row = idx / (kC * kP * kP);
    const int col = idx % (kC * kP * kP);
    const int b = row / kL, l = row % kL;
    const int py = l / (kIMG / kP), px = l % (kIMG / kP);
    const int c = col / (kP * kP), r = col % (kP * kP);
    const int iy = r / kP, ix = r % kP;
    A0[idx] = (_Float16)
        x[(((size_t)b * kC + c) * kIMG + py * kP + iy) * kIMG + px * kP + ix];
}

__global__ __launch_bounds__(256) void add_pos_kernel(
    float* __restrict__ h, const float* __restrict__ pos)
{
    const int idx = blockIdx.x * blockDim.x + threadIdx.x;
    if (idx >= kTOK * kD) return;
    const int l = (idx / kD) % kL;
    const int d = idx % kD;
    h[idx] += pos[(size_t)l * kD + d];
}

// ---------------------------------------------------------------------------
// LayerNorm: one wave32 per token; optional fp32 and fp16 outputs.
// ---------------------------------------------------------------------------
__global__ __launch_bounds__(256) void layernorm_kernel(
    const float* __restrict__ x, const float* __restrict__ g,
    const float* __restrict__ b, float* __restrict__ out,
    _Float16* __restrict__ out16, int ntok)
{
    const int wave = threadIdx.x >> 5;
    const int lane = threadIdx.x & 31;
    const int tok = blockIdx.x * 8 + wave;
    if (tok >= ntok) return;
    const float* xp = x + (size_t)tok * kD;
    float s = 0.f, s2 = 0.f;
    for (int i = lane; i < kD; i += 32) {
        const float v = xp[i];
        s += v; s2 += v * v;
    }
    #pragma unroll
    for (int off = 16; off > 0; off >>= 1) {
        s  += __shfl_xor(s,  off, 32);
        s2 += __shfl_xor(s2, off, 32);
    }
    const float mu  = s / kD;
    const float var = s2 / kD - mu * mu;
    const float rs  = rsqrtf(var + 1e-5f);
    for (int i = lane; i < kD; i += 32) {
        const float v = (xp[i] - mu) * rs * g[i] + b[i];
        if (out)   out[(size_t)tok * kD + i] = v;
        if (out16) out16[(size_t)tok * kD + i] = (_Float16)v;
    }
}

// ---------------------------------------------------------------------------
// Depthwise causal conv (K=4) over L + SiLU; writes fp32 + fp16.
// ---------------------------------------------------------------------------
__global__ __launch_bounds__(256) void conv_silu_kernel(
    const float* __restrict__ xz, const float* __restrict__ cw,
    const float* __restrict__ cb, float* __restrict__ xc,
    _Float16* __restrict__ xc16)
{
    const int idx = blockIdx.x * blockDim.x + threadIdx.x;
    if (idx >= kTOK * kDI) return;
    const int di = idx % kDI;
    const int l  = (idx / kDI) % kL;
    const int b  = idx / (kDI * kL);
    float acc = cb[di];
    #pragma unroll
    for (int k = 0; k < kK; ++k) {
        const int ls = l - (kK - 1) + k;
        if (ls >= 0)
            acc += cw[di * kK + k] * xz[((size_t)(b * kL + ls)) * kXZ + di];
    }
    const float v = acc * (1.0f / (1.0f + __expf(-acc)));  // silu
    xc[idx] = v;
    xc16[idx] = (_Float16)v;
}

__global__ __launch_bounds__(256) void softplus_kernel(
    float* __restrict__ dt, const float* __restrict__ dtb)
{
    const int idx = blockIdx.x * blockDim.x + threadIdx.x;
    if (idx >= kTOK * kDI) return;
    const float v = dt[idx] + dtb[idx % kDI];
    dt[idx] = (v > 20.0f) ? v : log1pf(__expf(v));
}

// ---------------------------------------------------------------------------
// Selective scan: one thread per (b, di); N=16 states in registers; L=196.
// Writes gated output directly as fp16 (A operand of out-proj GEMM).
// ---------------------------------------------------------------------------
__global__ __launch_bounds__(256) void scan_kernel(
    const float* __restrict__ dt, const float* __restrict__ xc,
    const float* __restrict__ dbc, const float* __restrict__ xz,
    const float* __restrict__ A_log, const float* __restrict__ Dsk,
    _Float16* __restrict__ y16)
{
    const int idx = blockIdx.x * blockDim.x + threadIdx.x;
    if (idx >= kB * kDI) return;
    const int b = idx / kDI, di = idx % kDI;
    float Av[kN];
    #pragma unroll
    for (int n = 0; n < kN; ++n) Av[n] = -__expf(A_log[(size_t)di * kN + n]);
    const float dsk = Dsk[di];
    float h[kN];
    #pragma unroll
    for (int n = 0; n < kN; ++n) h[n] = 0.0f;

    for (int l = 0; l < kL; ++l) {
        const size_t row = (size_t)(b * kL + l);
        const float dtv = dt[row * kDI + di];
        const float xv  = xc[row * kDI + di];
        const float* dr = dbc + row * kDBC;
        float acc = 0.0f;
        #pragma unroll
        for (int n = 0; n < kN; ++n) {
            const float da = __expf(dtv * Av[n]);
            h[n] = da * h[n] + dtv * dr[kR + n] * xv;
            acc += h[n] * dr[kR + kN + n];
        }
        acc += dsk * xv;
        const float z = xz[row * kXZ + kDI + di];
        acc *= z * (1.0f / (1.0f + __expf(-z)));      // y * silu(z)
        y16[row * kDI + di] = (_Float16)acc;
    }
}

// mean over L -> fp16 feature matrix padded to 128 rows (head GEMM A operand)
__global__ __launch_bounds__(256) void meanpool_f16_kernel(
    const float* __restrict__ hN, _Float16* __restrict__ feat16)
{
    const int idx = blockIdx.x * blockDim.x + threadIdx.x;
    if (idx >= 128 * kD) return;
    const int b = idx / kD, d = idx % kD;
    float s = 0.f;
    if (b < kB) {
        for (int l = 0; l < kL; ++l) s += hN[((size_t)b * kL + l) * kD + d];
        s /= (float)kL;
    }
    feat16[idx] = (_Float16)s;
}

// ---------------------------------------------------------------------------
// Host orchestration
// ---------------------------------------------------------------------------
static inline void launch_gemm(const _Float16* A, const _Float16* Bt,
                               float* C, _Float16* C16, const float* bias,
                               int M, int N, int K, int lda, int ldb,
                               int ldc, int ldc16, int beta, hipStream_t s)
{
    dim3 grid((M + TM - 1) / TM, (N + TN - 1) / TN);
    gemm_f16_wmma_kernel<<<grid, 256, 0, s>>>(A, Bt, C, C16, bias,
                                              M, N, K, lda, ldb, ldc, ldc16, beta);
}

extern "C" void kernel_launch(void* const* d_in, const int* in_sizes, int n_in,
                              void* d_out, int out_size, void* d_ws, size_t ws_size,
                              hipStream_t stream)
{
    const float* x       = (const float*)d_in[0];
    const float* patch_w = (const float*)d_in[1];
    const float* patch_b = (const float*)d_in[2];
    const float* pos     = (const float*)d_in[3];
    const float* bn_g    = (const float*)d_in[4];
    const float* bn_b    = (const float*)d_in[5];
    const float* in_w    = (const float*)d_in[6];
    const float* conv_w  = (const float*)d_in[7];
    const float* conv_b  = (const float*)d_in[8];
    const float* xp_w    = (const float*)d_in[9];
    const float* dtp_w   = (const float*)d_in[10];
    const float* dtp_b   = (const float*)d_in[11];
    const float* A_log   = (const float*)d_in[12];
    const float* D_skip  = (const float*)d_in[13];
    const float* out_w   = (const float*)d_in[14];
    const float* norm_g  = (const float*)d_in[15];
    const float* norm_b  = (const float*)d_in[16];
    const float* head_w  = (const float*)d_in[17];
    const float* head_b  = (const float*)d_in[18];
    float* out = (float*)d_out;

    // byte-based workspace carving, 256B aligned
    char* wsb = (char*)d_ws;
    auto carve = [&](size_t bytes) {
        char* p = wsb;
        wsb += (bytes + 255) & ~(size_t)255;
        return p;
    };
    // fp32 activations
    float* h    = (float*)carve((size_t)kTOK * kD * 4);
    float* hNf  = (float*)carve((size_t)kTOK * kD * 4);
    float* xz   = (float*)carve((size_t)kTOK * kXZ * 4);
    float* xc   = (float*)carve((size_t)kTOK * kDI * 4);
    float* dbc  = (float*)carve((size_t)kTOK * kDBC * 4);
    float* dtb  = (float*)carve((size_t)kTOK * kDI * 4);
    // fp16 GEMM operands (padded)
    _Float16* A016   = (_Float16*)carve((size_t)kTOK * kDI * 2);
    _Float16* hN16   = (_Float16*)carve((size_t)kTOK * kD * 2);
    _Float16* xc16   = (_Float16*)carve((size_t)kTOK * kDI * 2);
    _Float16* dbc16  = (_Float16*)carve((size_t)kTOK * 64 * 2);
    _Float16* y16    = (_Float16*)carve((size_t)kTOK * kDI * 2);
    _Float16* feat16 = (_Float16*)carve((size_t)128 * kD * 2);
    // fp16 weights, n-major, padded
    _Float16* wpatchT = (_Float16*)carve((size_t)kD * kDI * 2);             // [384][768]
    _Float16* winT    = (_Float16*)carve((size_t)kDEPTH * kXZ * kD * 2);    // [1536][384] x12
    _Float16* wxpT    = (_Float16*)carve((size_t)kDEPTH * 64 * kDI * 2);    // [64][768] x12
    _Float16* wdtpT   = (_Float16*)carve((size_t)kDEPTH * kDI * 32 * 2);    // [768][32] x12
    _Float16* woutT   = (_Float16*)carve((size_t)kDEPTH * kD * kDI * 2);    // [384][768] x12
    _Float16* wheadT  = (_Float16*)carve((size_t)1024 * kD * 2);            // [1024][384]

    auto blocks = [](long n) { return (int)((n + 255) / 256); };
    const int T = 256;

    // ---- one-time weight conversion / transposition to fp16 n-major ----
    // patch_w is already [N=384][K=768]
    convert_copy_f16_kernel<<<blocks((long)kD * kDI), T, 0, stream>>>(
        patch_w, wpatchT, (long)kD * kDI);
    convert_transpose_f16_kernel<<<blocks((long)1024 * kD), T, 0, stream>>>(
        head_w, wheadT, kD, kNC, kD, 1024);
    for (int dep = 0; dep < kDEPTH; ++dep) {
        convert_transpose_f16_kernel<<<blocks((long)kXZ * kD), T, 0, stream>>>(
            in_w + (size_t)dep * kD * kXZ, winT + (size_t)dep * kXZ * kD,
            kD, kXZ, kD, kXZ);
        convert_transpose_f16_kernel<<<blocks((long)64 * kDI), T, 0, stream>>>(
            xp_w + (size_t)dep * kDI * kDBC, wxpT + (size_t)dep * 64 * kDI,
            kDI, kDBC, kDI, 64);
        convert_transpose_f16_kernel<<<blocks((long)kDI * 32), T, 0, stream>>>(
            dtp_w + (size_t)dep * kR * kDI, wdtpT + (size_t)dep * kDI * 32,
            kR, kDI, 32, kDI);
        convert_transpose_f16_kernel<<<blocks((long)kD * kDI), T, 0, stream>>>(
            out_w + (size_t)dep * kDI * kD, woutT + (size_t)dep * kD * kDI,
            kDI, kD, kDI, kD);
    }

    // ---- patch embedding ----
    im2col_f16_kernel<<<blocks((long)kTOK * kDI), T, 0, stream>>>(x, A016);
    launch_gemm(A016, wpatchT, h, nullptr, patch_b,
                kTOK, kD, kDI, kDI, kDI, kD, 0, 0, stream);
    add_pos_kernel<<<blocks((long)kTOK * kD), T, 0, stream>>>(h, pos);

    // ---- mamba blocks ----
    for (int dep = 0; dep < kDEPTH; ++dep) {
        layernorm_kernel<<<(kTOK + 7) / 8, T, 0, stream>>>(
            h, bn_g + (size_t)dep * kD, bn_b + (size_t)dep * kD,
            nullptr, hN16, kTOK);
        // xz = hN @ in_w                       (6272 x 384 x 1536)
        launch_gemm(hN16, winT + (size_t)dep * kXZ * kD, xz, nullptr, nullptr,
                    kTOK, kXZ, kD, kD, kD, kXZ, 0, 0, stream);
        conv_silu_kernel<<<blocks((long)kTOK * kDI), T, 0, stream>>>(
            xz, conv_w + (size_t)dep * kDI * kK, conv_b + (size_t)dep * kDI,
            xc, xc16);
        // dbc = xc @ xp_w                      (6272 x 768 x 56)
        launch_gemm(xc16, wxpT + (size_t)dep * 64 * kDI, dbc, dbc16, nullptr,
                    kTOK, kDBC, kDI, kDI, kDI, kDBC, 64, 0, stream);
        // dt = dbc[:, :24] @ dtp_w             (K padded to 32; B pad rows = 0)
        launch_gemm(dbc16, wdtpT + (size_t)dep * kDI * 32, dtb, nullptr, nullptr,
                    kTOK, kDI, 32, 64, 32, kDI, 0, 0, stream);
        softplus_kernel<<<blocks((long)kTOK * kDI), T, 0, stream>>>(
            dtb, dtp_b + (size_t)dep * kDI);
        scan_kernel<<<blocks((long)kB * kDI), T, 0, stream>>>(
            dtb, xc, dbc, xz, A_log + (size_t)dep * kDI * kN,
            D_skip + (size_t)dep * kDI, y16);
        // h += y @ out_w                       (6272 x 768 x 384)
        launch_gemm(y16, woutT + (size_t)dep * kD * kDI, h, nullptr, nullptr,
                    kTOK, kD, kDI, kDI, kDI, kD, 0, 1, stream);
    }

    // ---- final LN, mean pool, head ----
    layernorm_kernel<<<(kTOK + 7) / 8, T, 0, stream>>>(
        h, norm_g, norm_b, hNf, nullptr, kTOK);
    meanpool_f16_kernel<<<blocks((long)128 * kD), T, 0, stream>>>(hNf, feat16);
    launch_gemm(feat16, wheadT, out, nullptr, head_b,
                kB, kNC, kD, kD, kD, kNC, 0, 0, stream);
}
